// myLSTM_76209899701130
// MI455X (gfx1250) — matrix operand order repaired and checked by reading, not compiled, not run on hardware
//
#include <hip/hip_runtime.h>
#include <math.h>

typedef __attribute__((ext_vector_type(16))) __bf16 v16bf;
typedef __attribute__((ext_vector_type(8)))  float  v8f;
typedef unsigned short u16;
typedef unsigned int   u32;

#define T_STEPS 36
#define BATCH   128
#define DIN     4096
#define HID     2048
#define H3      (3*HID)

// ---------- helpers ----------
__device__ __forceinline__ u16 f2bf(float f) {
    u32 u = __float_as_uint(f);
    u32 r = u + 0x7FFFu + ((u >> 16) & 1u);   // round-to-nearest-even
    return (u16)(r >> 16);
}
__device__ __forceinline__ float sigmoidf(float x) {
    return 1.0f / (1.0f + __expf(-x));
}

union FragAB { uint4 q[2]; v16bf v; };

// ---------- conversion kernels ----------
// feature [B,T,D] f32  ->  featbf [T*B, D] bf16  (transpose t<->b)
__global__ void cvt_feature(const float* __restrict__ in, u16* __restrict__ out) {
    const size_t total = (size_t)T_STEPS * BATCH * DIN;
    for (size_t i = (size_t)blockIdx.x * blockDim.x + threadIdx.x; i < total;
         i += (size_t)gridDim.x * blockDim.x) {
        size_t d   = i % DIN;
        size_t row = i / DIN;          // row = t*B + b
        size_t b   = row % BATCH;
        size_t t   = row / BATCH;
        out[i] = f2bf(in[(b * T_STEPS + t) * DIN + d]);
    }
}

__global__ void cvt_bf16(const float* __restrict__ in, u16* __restrict__ out, size_t n) {
    for (size_t i = (size_t)blockIdx.x * blockDim.x + threadIdx.x; i < n;
         i += (size_t)gridDim.x * blockDim.x)
        out[i] = f2bf(in[i]);
}

__global__ void init_h(float* __restrict__ hcur, u16* __restrict__ hbf, int n) {
    int i = blockIdx.x * blockDim.x + threadIdx.x;
    if (i < n) { hcur[i] = 0.0f; hbf[i] = 0; }
}

// ---------- bf16 WMMA GEMM:  C[M,N] = A[M,K] * W[N,K]^T  ----------
// A row-major bf16, W row-major bf16 (W[n][k]), C row-major f32.
// Block tile 128x128, K-step 32, 8 waves (wave32), wave tile 64Mx32N.
// Double-buffered LDS; staging via CDNA5 async copies (global -> LDS direct,
// tracked by ASYNCcnt), overlapped with the WMMA work of the current tile.
#define TM 128
#define TN 128
#define TK 32
#define LDSS 40   // ushorts per LDS row: 32 data + 8 pad (80B stride -> conflict-free)

__device__ __forceinline__ void async_b128(unsigned lds_off, unsigned byte_off,
                                           const u16* base) {
    // GLOBAL_LOAD_ASYNC_TO_LDS_B128, GVS mode: saddr base + 32-bit vgpr offset.
    asm volatile("global_load_async_to_lds_b128 %0, %1, %2 offset:0"
                 :: "v"(lds_off), "v"(byte_off), "s"(base)
                 : "memory");
}
__device__ __forceinline__ void wait_async0() {
    asm volatile("s_wait_asynccnt 0x0" ::: "memory");
}

__global__ __launch_bounds__(256)
void gemm_bf16_nt(const u16* __restrict__ A, const u16* __restrict__ W,
                  float* __restrict__ C, int M, int N, int K)
{
    __shared__ __align__(16) u16 sA[2][TM * LDSS];
    __shared__ __align__(16) u16 sB[2][TN * LDSS];

    const int tid   = threadIdx.x;
    const int lane  = tid & 31;
    const int wave  = tid >> 5;            // 0..7
    const int bm    = blockIdx.y * TM;
    const int bn    = blockIdx.x * TN;
    const int wm    = (wave >> 2) * 64;    // 2 wave-rows
    const int wn    = (wave & 3)  * 32;    // 4 wave-cols
    const int lhalf = lane >> 4;           // 0|1
    const int lmod  = lane & 15;

    // Per-thread staging assignment: 2 chunks each of A and B per tile.
    const int r0 = tid >> 2;               // rows handled: r0, r0+64
    const int c16 = tid & 3;               // 16B chunk within the 64B row

    v8f zero;
    #pragma unroll
    for (int r = 0; r < 8; ++r) zero[r] = 0.0f;

    v8f acc[4][2];
    #pragma unroll
    for (int i = 0; i < 4; ++i)
        #pragma unroll
        for (int j = 0; j < 2; ++j) acc[i][j] = zero;

    // ---- stage first tile into buffer 0 (async, direct to LDS) ----
    {
        #pragma unroll
        for (int i = 0; i < 2; ++i) {
            int row = r0 + i * 64;
            unsigned la = (unsigned)(size_t)&sA[0][row * LDSS + c16 * 8];
            unsigned ga = (unsigned)(((size_t)(bm + row)) * K + c16 * 8) * 2u;
            async_b128(la, ga, A);
            unsigned lb = (unsigned)(size_t)&sB[0][row * LDSS + c16 * 8];
            unsigned gb = (unsigned)(((size_t)(bn + row)) * K + c16 * 8) * 2u;
            async_b128(lb, gb, W);
        }
    }
    wait_async0();
    __syncthreads();

    for (int k0 = 0; k0 < K; k0 += TK) {
        const int cur = (k0 / TK) & 1;

        // ---- issue async staging of the NEXT tile into the other buffer ----
        if (k0 + TK < K) {
            const int nxt = cur ^ 1;
            #pragma unroll
            for (int i = 0; i < 2; ++i) {
                int row = r0 + i * 64;
                unsigned la = (unsigned)(size_t)&sA[nxt][row * LDSS + c16 * 8];
                unsigned ga = (unsigned)(((size_t)(bm + row)) * K + (k0 + TK) + c16 * 8) * 2u;
                async_b128(la, ga, A);
                unsigned lb = (unsigned)(size_t)&sB[nxt][row * LDSS + c16 * 8];
                unsigned gb = (unsigned)(((size_t)(bn + row)) * K + (k0 + TK) + c16 * 8) * 2u;
                async_b128(lb, gb, W);
            }
        }

        // ---- compute on the current buffer ----
        const u16* cA = &sA[cur][0];
        const u16* cB = &sB[cur][0];

        // B fragments (32x16 bf16): lane -> col n=lmod, k = 0..15 | 16..31 by half-wave.
        FragAB bfr[2];
        #pragma unroll
        for (int j = 0; j < 2; ++j) {
            const u16* p = &cB[(wn + j * 16 + lmod) * LDSS + lhalf * 16];
            bfr[j].q[0] = *reinterpret_cast<const uint4*>(p);
            bfr[j].q[1] = *reinterpret_cast<const uint4*>(p + 8);
        }
        // A fragments (16x32 bf16): lane -> row m=lmod, k = 0..15 | 16..31 by half-wave.
        #pragma unroll
        for (int i = 0; i < 4; ++i) {
            const u16* p = &cA[(wm + i * 16 + lmod) * LDSS + lhalf * 16];
            FragAB afr;
            afr.q[0] = *reinterpret_cast<const uint4*>(p);
            afr.q[1] = *reinterpret_cast<const uint4*>(p + 8);
            #pragma unroll
            for (int j = 0; j < 2; ++j) {
                acc[i][j] = __builtin_amdgcn_wmma_f32_16x16x32_bf16(
                    false, afr.v, false, bfr[j].v, (short)0, acc[i][j], false, false);
            }
        }

        // Next tile fully in LDS (our own async ops), then block-wide sync.
        wait_async0();
        __syncthreads();
    }

    // D layout: VGPR r -> (m = r + 8*lhalf, n = lmod)
    #pragma unroll
    for (int i = 0; i < 4; ++i)
        #pragma unroll
        for (int j = 0; j < 2; ++j)
            #pragma unroll
            for (int r = 0; r < 8; ++r) {
                int m = bm + wm + i * 16 + r + lhalf * 8;
                int n = bn + wn + j * 16 + lmod;
                C[(size_t)m * N + n] = acc[i][j][r];
            }
}

// ---------- fused GRU gate/update kernel (one step) ----------
__global__ __launch_bounds__(256)
void gru_elem(const float* __restrict__ gi,   // [B,3H] slice of gis (no bias)
              const float* __restrict__ gh,   // [B,3H] (no bias)
              const float* __restrict__ bih, const float* __restrict__ bhh,
              float* __restrict__ hcur,       // [B,H] f32, in/out
              u16*  __restrict__ hbf,         // [B,H] bf16 for next GEMM
              float* __restrict__ hs_t)       // [B,H] f32 for heads
{
    int idx = blockIdx.x * blockDim.x + threadIdx.x;   // over B*H
    int b = idx / HID, j = idx % HID;
    size_t base = (size_t)b * H3;
    float ir = gi[base + j]           + bih[j];
    float hr = gh[base + j]           + bhh[j];
    float iz = gi[base + HID + j]     + bih[HID + j];
    float hz = gh[base + HID + j]     + bhh[HID + j];
    float in_ = gi[base + 2*HID + j]  + bih[2*HID + j];
    float hn  = gh[base + 2*HID + j]  + bhh[2*HID + j];
    float r = sigmoidf(ir + hr);
    float z = sigmoidf(iz + hz);
    float n = tanhf(in_ + r * hn);
    float h = hcur[idx];
    float hnew = (1.0f - z) * n + z * h;
    hcur[idx] = hnew;
    hs_t[idx] = hnew;
    hbf[idx]  = f2bf(hnew);
}

// ---------- heads: one wave per (t,b) row ----------
__global__ __launch_bounds__(256)
void heads_kernel(const float* __restrict__ hs,
                  const float* __restrict__ Wp, const float* __restrict__ bp,
                  const float* __restrict__ Wu, const float* __restrict__ bu,
                  float* __restrict__ out_hidden, float* __restrict__ out_util,
                  int rows)
{
    int gw   = blockIdx.x * (blockDim.x >> 5) + (threadIdx.x >> 5);
    int lane = threadIdx.x & 31;
    if (gw >= rows) return;
    const float* h = hs + (size_t)gw * HID;
    float a0 = 0.f, a1 = 0.f, au = 0.f;
    for (int j = lane; j < HID; j += 32) {
        float v = h[j];
        a0 += v * Wp[j];
        a1 += v * Wp[HID + j];
        au += v * Wu[j];
    }
    #pragma unroll
    for (int off = 16; off > 0; off >>= 1) {
        a0 += __shfl_down(a0, off, 32);
        a1 += __shfl_down(a1, off, 32);
        au += __shfl_down(au, off, 32);
    }
    if (lane == 0) {
        out_hidden[gw * 2 + 0] = a0 + bp[0];
        out_hidden[gw * 2 + 1] = a1 + bp[1];
        out_util[gw]           = au + bu[0];
    }
}

// ---------- workspace layout (bytes) ----------
#define OFF_FEATBF  ((size_t)0)                                   // T*B*DIN bf16
#define OFF_WIHBF   (OFF_FEATBF + (size_t)T_STEPS*BATCH*DIN*2)    // 3H*DIN bf16
#define OFF_WHHBF   (OFF_WIHBF  + (size_t)H3*DIN*2)               // 3H*H bf16
#define OFF_GIS     (OFF_WHHBF  + (size_t)H3*HID*2)               // T*B*3H f32
#define OFF_GH      (OFF_GIS    + (size_t)T_STEPS*BATCH*H3*4)     // B*3H f32
#define OFF_HCUR    (OFF_GH     + (size_t)BATCH*H3*4)             // B*H f32
#define OFF_HBF     (OFF_HCUR   + (size_t)BATCH*HID*4)            // B*H bf16
#define OFF_HS      (OFF_HBF    + (size_t)BATCH*HID*2)            // T*B*H f32

extern "C" void kernel_launch(void* const* d_in, const int* in_sizes, int n_in,
                              void* d_out, int out_size, void* d_ws, size_t ws_size,
                              hipStream_t stream) {
    const float* feature = (const float*)d_in[0];
    const float* W_ih    = (const float*)d_in[1];
    const float* W_hh    = (const float*)d_in[2];
    const float* b_ih    = (const float*)d_in[3];
    const float* b_hh    = (const float*)d_in[4];
    const float* W_pred  = (const float*)d_in[5];
    const float* b_pred  = (const float*)d_in[6];
    const float* W_util  = (const float*)d_in[7];
    const float* b_util  = (const float*)d_in[8];

    char* ws = (char*)d_ws;
    u16*   featbf = (u16*)  (ws + OFF_FEATBF);
    u16*   Wihbf  = (u16*)  (ws + OFF_WIHBF);
    u16*   Whhbf  = (u16*)  (ws + OFF_WHHBF);
    float* gis    = (float*)(ws + OFF_GIS);
    float* gh     = (float*)(ws + OFF_GH);
    float* hcur   = (float*)(ws + OFF_HCUR);
    u16*   hbf    = (u16*)  (ws + OFF_HBF);
    float* hs     = (float*)(ws + OFF_HS);

    float* out_hidden = (float*)d_out;                            // [T,B,2]
    float* out_util   = out_hidden + (size_t)T_STEPS * BATCH * 2; // [T,B,1]

    // 1) bf16 conversions + h0 = 0
    cvt_feature<<<8192, 256, 0, stream>>>(feature, featbf);
    cvt_bf16<<<8192, 256, 0, stream>>>(W_ih, Wihbf, (size_t)H3 * DIN);
    cvt_bf16<<<8192, 256, 0, stream>>>(W_hh, Whhbf, (size_t)H3 * HID);
    init_h<<<(BATCH * HID) / 256, 256, 0, stream>>>(hcur, hbf, BATCH * HID);

    // 2) big input-side GEMM: gis[T*B, 3H] = featbf * W_ih^T
    {
        dim3 grid(H3 / TN, (T_STEPS * BATCH) / TM);
        gemm_bf16_nt<<<grid, 256, 0, stream>>>(featbf, Wihbf, gis,
                                               T_STEPS * BATCH, H3, DIN);
    }

    // 3) recurrence: 36 sequential (GEMM + fused gates)
    for (int t = 0; t < T_STEPS; ++t) {
        dim3 grid(H3 / TN, BATCH / TM);   // (48, 1)
        gemm_bf16_nt<<<grid, 256, 0, stream>>>(hbf, Whhbf, gh, BATCH, H3, HID);
        gru_elem<<<(BATCH * HID) / 256, 256, 0, stream>>>(
            gis + (size_t)t * BATCH * H3, gh, b_ih, b_hh,
            hcur, hbf, hs + (size_t)t * BATCH * HID);
    }

    // 4) heads: hidden [T,B,2], utility [T,B,1]
    heads_kernel<<<(T_STEPS * BATCH) / 8, 256, 0, stream>>>(
        hs, W_pred, b_pred, W_util, b_util, out_hidden, out_util,
        T_STEPS * BATCH);
}
